// LSTMEncoder_58720792871304
// MI455X (gfx1250) — compile-verified, matrix operand order checked by reading, not run on hardware
//
#include <hip/hip_runtime.h>

// ---------------- CDNA5 WMMA types ----------------
typedef __bf16 v16bf __attribute__((ext_vector_type(16)));
typedef float  v8f   __attribute__((ext_vector_type(8)));

#define WMMA_BF16(A, B, C) \
  __builtin_amdgcn_wmma_f32_16x16x32_bf16(false, (A), false, (B), (short)0, (C), false, false)

// ---------------- problem dims ----------------
static constexpr int Sq = 128;          // seq len
static constexpr int Bb = 128;          // batch
static constexpr int E  = 256;          // embed
static constexpr int H  = 256;          // hidden
static constexpr int H2 = 512;          // 2*H
static constexpr int G4 = 1024;         // 4*H
static constexpr int MROWS = Bb * Sq;   // 16384

static constexpr int WPC  = 8;          // workgroups per LSTM chain
static constexpr int WSTR = 520;        // padded LDS row stride (ushorts) for W slice
static constexpr int XSTR = 264;        // padded LDS row stride (ushorts) for X tile

// ---------------- helpers ----------------
__device__ __forceinline__ unsigned short f2bf(float f) {
  unsigned int u = __float_as_uint(f);
  u += 0x7fffu + ((u >> 16) & 1u);      // round to nearest even
  return (unsigned short)(u >> 16);
}

// Fragment from row-major data: lane supplies the row pointer.
// Per ISA 7.12.2: element j -> K = k0 + (j<8 ? j : 16+j-8) + half*8
// => two contiguous 16-byte chunks at k0+half*8 and k0+16+half*8.
__device__ __forceinline__ v16bf ld_frag_row(const unsigned short* row, int k0, int half) {
  union { v16bf v; int4 q[2]; } u;
  const unsigned short* p = row + k0 + half * 8;
  u.q[0] = *(const int4*)(p);
  u.q[1] = *(const int4*)(p + 16);
  return u.v;
}

// B fragment from row-major [K][N] data (A@B case): strided gather.
__device__ __forceinline__ v16bf ld_frag_kn(const unsigned short* base, int ld, int k0, int n, int half) {
  union { v16bf v; unsigned short u[16]; } r;
#pragma unroll
  for (int j = 0; j < 8; ++j) {
    r.u[j]     = base[(k0 + half * 8 + j) * ld + n];
    r.u[j + 8] = base[(k0 + 16 + half * 8 + j) * ld + n];
  }
  return r.v;
}

__device__ __forceinline__ float wred_max(float x) {
#pragma unroll
  for (int o = 16; o > 0; o >>= 1) x = fmaxf(x, __shfl_xor(x, o, 32));
  return x;
}
__device__ __forceinline__ float wred_sum(float x) {
#pragma unroll
  for (int o = 16; o > 0; o >>= 1) x += __shfl_xor(x, o, 32);
  return x;
}
__device__ __forceinline__ float sigm(float x) { return 1.0f / (1.0f + __expf(-x)); }

__device__ __forceinline__ void async_b128(unsigned int lds, unsigned long long ga) {
  asm volatile("global_load_async_to_lds_b128 %0, %1, off" :: "v"(lds), "v"(ga) : "memory");
}
__device__ __forceinline__ void wait_async0() {
  asm volatile("s_wait_asynccnt 0x0" ::: "memory");
}

// ---------------- prep: bf16 weights, combined [Wih|Whh], summed bias ----------------
__global__ void __launch_bounds__(256) prep_kernel(
    const float* __restrict__ Wihf, const float* __restrict__ Whhf,
    const float* __restrict__ Wihb, const float* __restrict__ Whhb,
    const float* __restrict__ bihf, const float* __restrict__ bhhf,
    const float* __restrict__ bihb, const float* __restrict__ bhhb,
    const float* __restrict__ S1W,
    unsigned short* __restrict__ Wcomb,   // [2][1024][512] bf16
    float* __restrict__ bias2,            // [2][1024]
    unsigned short* __restrict__ S1Wb) {  // [512][512] bf16
  int tid = blockIdx.x * blockDim.x + threadIdx.x;
  int stride = gridDim.x * blockDim.x;
  const int NW = 2 * G4 * H2;
  for (int i = tid; i < NW; i += stride) {
    int d = i / (G4 * H2);
    int rem = i - d * (G4 * H2);
    int n = rem / H2, k = rem - n * H2;
    const float* Wi = d ? Wihb : Wihf;
    const float* Wh = d ? Whhb : Whhf;
    float v = (k < E) ? Wi[n * E + k] : Wh[n * E + (k - E)];
    Wcomb[i] = f2bf(v);
  }
  for (int i = tid; i < H2 * H2; i += stride) S1Wb[i] = f2bf(S1W[i]);
  for (int i = tid; i < 2 * G4; i += stride) {
    int d = i >> 10, n = i & 1023;
    bias2[i] = d ? (bihb[n] + bhhb[n]) : (bihf[n] + bhhf[n]);
  }
}

// ---------------- embedding gather -> bf16 X[input][b*S+s][e] ----------------
__global__ void __launch_bounds__(256) gather_kernel(
    const int* __restrict__ iq, const int* __restrict__ ic, const int* __restrict__ ic2,
    const float* __restrict__ emb, unsigned short* __restrict__ Xall) {
  long total = 3L * MROWS * E;
  for (long i = (long)blockIdx.x * blockDim.x + threadIdx.x; i < total;
       i += (long)gridDim.x * blockDim.x) {
    int e = (int)(i & (E - 1));
    long rowi = i >> 8;
    int input = (int)(rowi / MROWS);
    int row = (int)(rowi - (long)input * MROWS);  // b*S + s
    int b = row >> 7, s = row & (Sq - 1);
    const int* idx = input == 0 ? iq : (input == 1 ? ic : ic2);
    int v = idx[s * Bb + b];
    Xall[i] = f2bf(emb[(long)v * E + e]);
  }
}

// ---------------- weight-stationary BiLSTM ----------------
// grid = 6 chains * 8 WGs, block = 256 (8 waves).
// WG j of a chain owns hidden cols [j*32, j*32+32): its 128KB weight slice lives in
// LDS for the whole kernel. h is double-buffered in global (L2); a software chain
// barrier (atomic counter + s_sleep poll) synchronizes the 8 WGs per timestep.
// X_t tiles are async-prefetched into double-buffered LDS. c stays in VGPRs.
__global__ void __launch_bounds__(256) lstm_chain_kernel(
    const unsigned short* __restrict__ Xall,   // [3][16384][256] bf16
    const unsigned short* __restrict__ Wcomb,  // [2][1024][512] bf16
    const float* __restrict__ bias2,           // [2][1024]
    const float* __restrict__ hq, const float* __restrict__ cq,
    const float* __restrict__ hc, const float* __restrict__ cc,
    const float* __restrict__ hc2, const float* __restrict__ cc2,
    unsigned short* __restrict__ hg,           // [6][2][128][256] bf16
    unsigned int* __restrict__ bar,            // [6][16]
    unsigned short* __restrict__ Yall) {       // [3][16384][512] bf16
  extern __shared__ unsigned short smem[];
  unsigned short* wlds = smem;                 // [128][WSTR]  (gate*32+localcol rows)
  unsigned short* xb0 = smem + 128 * WSTR;     // [128][XSTR]
  unsigned short* xb1 = xb0 + 128 * XSTR;

  int chain = blockIdx.x >> 3;                 // 0..5
  int j = blockIdx.x & 7;                      // WG within chain
  int input = chain >> 1, dir = chain & 1;
  int jc0 = j * 32;                            // first hidden col owned

  const unsigned short* X = Xall + (size_t)input * MROWS * E;
  const unsigned short* Wg = Wcomb + (size_t)dir * G4 * H2;
  const float* h0 = input == 0 ? hq : (input == 1 ? hc : hc2);
  const float* c0 = input == 0 ? cq : (input == 1 ? cc : cc2);
  unsigned short* Y = Yall + (size_t)input * MROWS * H2;
  unsigned short* hgc = hg + (size_t)chain * 2 * (Bb * H);
  unsigned int* ctr = bar + chain * 16;

  int tid = threadIdx.x;
  int lane = tid & 31, wid = tid >> 5;
  int half = lane >> 4, l15 = lane & 15;
  int wm = wid >> 1, wn = wid & 1;             // waves: 4 M-groups x 2 col-halves
  int colL = wn * 16 + l15;                    // local col 0..31
  int colG = jc0 + colL;                       // global hidden col

  // ---- stage weight slice into LDS (async), 8192 x 16B chunks ----
#pragma unroll
  for (int i = 0; i < 32; ++i) {
    int c = i * 256 + tid;
    int R = c >> 6, off = c & 63;              // R = gate*32 + localrow
    int g = R >> 5, lr = R & 31;
    const unsigned short* gp = Wg + (size_t)(g * H + jc0 + lr) * H2 + off * 8;
    async_b128((unsigned int)(size_t)(wlds + R * WSTR + off * 8),
               (unsigned long long)(size_t)gp);
  }
  // ---- stage X for first timestep (async), 4096 x 16B chunks ----
  int ts0 = dir ? (Sq - 1) : 0;
#pragma unroll
  for (int i = 0; i < 16; ++i) {
    int c = i * 256 + tid;
    int row = c >> 5, chunk = c & 31;
    const unsigned short* gp = X + ((size_t)row * Sq + ts0) * E + chunk * 8;
    async_b128((unsigned int)(size_t)(xb0 + row * XSTR + chunk * 8),
               (unsigned long long)(size_t)gp);
  }

  // ---- init h (this WG writes 1/8 of it) and c registers ----
  for (int k = 0; k < 16; ++k) {
    int i = j * 4096 + k * 256 + tid;
    hgc[i] = f2bf(h0[(size_t)dir * Bb * H + i]);
  }
  float creg[2][8];
#pragma unroll
  for (int mt = 0; mt < 2; ++mt)
#pragma unroll
    for (int r = 0; r < 8; ++r) {
      int b = (wm * 2 + mt) * 16 + r + half * 8;
      creg[mt][r] = c0[(size_t)dir * Bb * H + b * H + colG];
    }
  float bsr[4];
#pragma unroll
  for (int g = 0; g < 4; ++g) bsr[g] = bias2[dir * G4 + g * H + colG];

  const unsigned short* wr[4];
#pragma unroll
  for (int g = 0; g < 4; ++g) wr[g] = wlds + (size_t)(g * 32 + colL) * WSTR;

  wait_async0();
  // ---- chain barrier (phase 1): h init + LDS staging visible ----
  int phase = 1;
  __syncthreads();
  if (tid == 0) {
    __threadfence();
    atomicAdd(ctr, 1u);
    volatile unsigned int* vc = (volatile unsigned int*)ctr;
    while (*vc < (unsigned)(WPC * phase)) __builtin_amdgcn_s_sleep(2);
  }
  __syncthreads();
  __threadfence();

  for (int t = 0; t < Sq; ++t) {
    const unsigned short* xR = (t & 1) ? xb1 : xb0;
    unsigned short* xN = (t & 1) ? xb0 : xb1;
    const unsigned short* hgR = hgc + (size_t)(t & 1) * (Bb * H);
    unsigned short* hgW = hgc + (size_t)((t + 1) & 1) * (Bb * H);
    int ts = dir ? (Sq - 1 - t) : t;

    if (t + 1 < Sq) {
      int tsn = dir ? (Sq - 2 - t) : (t + 1);
#pragma unroll
      for (int i = 0; i < 16; ++i) {
        int c = i * 256 + tid;
        int row = c >> 5, chunk = c & 31;
        const unsigned short* gp = X + ((size_t)row * Sq + tsn) * E + chunk * 8;
        async_b128((unsigned int)(size_t)(xN + row * XSTR + chunk * 8),
                   (unsigned long long)(size_t)gp);
      }
    }

    v8f acc[2][4];
#pragma unroll
    for (int mt = 0; mt < 2; ++mt)
#pragma unroll
      for (int g = 0; g < 4; ++g) {
        v8f z = {0.f, 0.f, 0.f, 0.f, 0.f, 0.f, 0.f, 0.f};
        acc[mt][g] = z;
      }

    // K = 0..255 : embedded input (LDS, async-prefetched)
#pragma unroll
    for (int kk = 0; kk < 8; ++kk) {
      int k0 = kk * 32;
      v16bf w0 = ld_frag_row(wr[0], k0, half);
      v16bf w1 = ld_frag_row(wr[1], k0, half);
      v16bf w2 = ld_frag_row(wr[2], k0, half);
      v16bf w3 = ld_frag_row(wr[3], k0, half);
#pragma unroll
      for (int mt = 0; mt < 2; ++mt) {
        int aRow = (wm * 2 + mt) * 16 + l15;
        v16bf a = ld_frag_row(xR + aRow * XSTR, k0, half);
        acc[mt][0] = WMMA_BF16(a, w0, acc[mt][0]);
        acc[mt][1] = WMMA_BF16(a, w1, acc[mt][1]);
        acc[mt][2] = WMMA_BF16(a, w2, acc[mt][2]);
        acc[mt][3] = WMMA_BF16(a, w3, acc[mt][3]);
      }
    }
    // K = 256..511 : recurrent h (global, L2-resident)
#pragma unroll
    for (int kk = 0; kk < 8; ++kk) {
      int k0 = kk * 32;
      v16bf w0 = ld_frag_row(wr[0] + E, k0, half);
      v16bf w1 = ld_frag_row(wr[1] + E, k0, half);
      v16bf w2 = ld_frag_row(wr[2] + E, k0, half);
      v16bf w3 = ld_frag_row(wr[3] + E, k0, half);
#pragma unroll
      for (int mt = 0; mt < 2; ++mt) {
        int aRow = (wm * 2 + mt) * 16 + l15;
        v16bf a = ld_frag_row(hgR + (size_t)aRow * H, k0, half);
        acc[mt][0] = WMMA_BF16(a, w0, acc[mt][0]);
        acc[mt][1] = WMMA_BF16(a, w1, acc[mt][1]);
        acc[mt][2] = WMMA_BF16(a, w2, acc[mt][2]);
        acc[mt][3] = WMMA_BF16(a, w3, acc[mt][3]);
      }
    }

    // ---- gates, cell update, h out ----
#pragma unroll
    for (int mt = 0; mt < 2; ++mt)
#pragma unroll
      for (int r = 0; r < 8; ++r) {
        int b = (wm * 2 + mt) * 16 + r + half * 8;
        float ig = sigm(acc[mt][0][r] + bsr[0]);
        float fg = sigm(acc[mt][1][r] + bsr[1]);
        float gg = tanhf(acc[mt][2][r] + bsr[2]);
        float og = sigm(acc[mt][3][r] + bsr[3]);
        float cn = fg * creg[mt][r] + ig * gg;
        creg[mt][r] = cn;
        float hv = og * tanhf(cn);
        unsigned short hb = f2bf(hv);
        hgW[(size_t)b * H + colG] = hb;
        Y[((size_t)b * Sq + ts) * H2 + dir * H + colG] = hb;
      }

    wait_async0();
    // ---- chain barrier ----
    ++phase;
    __syncthreads();
    if (tid == 0) {
      __threadfence();
      atomicAdd(ctr, 1u);
      volatile unsigned int* vc = (volatile unsigned int*)ctr;
      while (*vc < (unsigned)(WPC * phase)) __builtin_amdgcn_s_sleep(2);
    }
    __syncthreads();
    __threadfence();
  }
}

// ---------------- Ct = tanh(C0 @ S1W^T + b), M=16384 N=512 K=512 ----------------
__global__ void __launch_bounds__(256) s1_gemm(
    const unsigned short* __restrict__ C0, const unsigned short* __restrict__ W,
    const float* __restrict__ bvec, unsigned short* __restrict__ Ct) {
  int lane = threadIdx.x & 31, wid = threadIdx.x >> 5;
  int half = lane >> 4, l15 = lane & 15;
  int wm = wid >> 1, wn = wid & 1;
  int mBase = blockIdx.x * 128 + wm * 32;
  int nBase = blockIdx.y * 64 + wn * 32;
  v8f acc[2][2];
#pragma unroll
  for (int i = 0; i < 2; ++i)
#pragma unroll
    for (int j = 0; j < 2; ++j) {
      v8f z = {0.f, 0.f, 0.f, 0.f, 0.f, 0.f, 0.f, 0.f};
      acc[i][j] = z;
    }
  const unsigned short* a0r = C0 + (size_t)(mBase + l15) * H2;
  const unsigned short* a1r = C0 + (size_t)(mBase + 16 + l15) * H2;
  const unsigned short* b0r = W + (size_t)(nBase + l15) * H2;
  const unsigned short* b1r = W + (size_t)(nBase + 16 + l15) * H2;
#pragma unroll 4
  for (int kk = 0; kk < 16; ++kk) {
    int k0 = kk * 32;
    v16bf a0 = ld_frag_row(a0r, k0, half);
    v16bf a1 = ld_frag_row(a1r, k0, half);
    v16bf b0 = ld_frag_row(b0r, k0, half);
    v16bf b1 = ld_frag_row(b1r, k0, half);
    acc[0][0] = WMMA_BF16(a0, b0, acc[0][0]);
    acc[0][1] = WMMA_BF16(a0, b1, acc[0][1]);
    acc[1][0] = WMMA_BF16(a1, b0, acc[1][0]);
    acc[1][1] = WMMA_BF16(a1, b1, acc[1][1]);
  }
#pragma unroll
  for (int i = 0; i < 2; ++i)
#pragma unroll
    for (int j = 0; j < 2; ++j)
#pragma unroll
      for (int r = 0; r < 8; ++r) {
        int m = mBase + i * 16 + r + half * 8;
        int n = nBase + j * 16 + l15;
        Ct[(size_t)m * H2 + n] = f2bf(tanhf(acc[i][j][r] + bvec[n]));
      }
}

// ---------------- Hm[z] = Q[z] @ Ct[z]^T, per batch: M=128 N=128 K=512 ----------------
__global__ void __launch_bounds__(256) hm_gemm(
    const unsigned short* __restrict__ Yq, const unsigned short* __restrict__ Ct,
    float* __restrict__ Hm) {
  int z = blockIdx.z;
  int lane = threadIdx.x & 31, wid = threadIdx.x >> 5;
  int half = lane >> 4, l15 = lane & 15;
  int wm = wid >> 1, wn = wid & 1;
  int mBase = wm * 32;
  int nBase = blockIdx.y * 64 + wn * 32;
  v8f acc[2][2];
#pragma unroll
  for (int i = 0; i < 2; ++i)
#pragma unroll
    for (int j = 0; j < 2; ++j) {
      v8f z8 = {0.f, 0.f, 0.f, 0.f, 0.f, 0.f, 0.f, 0.f};
      acc[i][j] = z8;
    }
  const unsigned short* a0r = Yq + (size_t)(z * Sq + mBase + l15) * H2;
  const unsigned short* a1r = Yq + (size_t)(z * Sq + mBase + 16 + l15) * H2;
  const unsigned short* b0r = Ct + (size_t)(z * Sq + nBase + l15) * H2;
  const unsigned short* b1r = Ct + (size_t)(z * Sq + nBase + 16 + l15) * H2;
#pragma unroll 4
  for (int kk = 0; kk < 16; ++kk) {
    int k0 = kk * 32;
    v16bf a0 = ld_frag_row(a0r, k0, half);
    v16bf a1 = ld_frag_row(a1r, k0, half);
    v16bf b0 = ld_frag_row(b0r, k0, half);
    v16bf b1 = ld_frag_row(b1r, k0, half);
    acc[0][0] = WMMA_BF16(a0, b0, acc[0][0]);
    acc[0][1] = WMMA_BF16(a0, b1, acc[0][1]);
    acc[1][0] = WMMA_BF16(a1, b0, acc[1][0]);
    acc[1][1] = WMMA_BF16(a1, b1, acc[1][1]);
  }
#pragma unroll
  for (int i = 0; i < 2; ++i)
#pragma unroll
    for (int j = 0; j < 2; ++j)
#pragma unroll
      for (int r = 0; r < 8; ++r) {
        int m = mBase + i * 16 + r + half * 8;
        int n = nBase + j * 16 + l15;
        Hm[(size_t)z * (Sq * Sq) + (size_t)m * Sq + n] = acc[i][j][r];
      }
}

// ---------------- softmax over last axis (AQ) ----------------
__global__ void __launch_bounds__(128) softmax_row(
    const float* __restrict__ Hm, unsigned short* __restrict__ AQ) {
  int row = blockIdx.x * 4 + (threadIdx.x >> 5);  // b*128 + s
  int lane = threadIdx.x & 31;
  const float* r = Hm + (size_t)row * Sq;
  float v[4];
#pragma unroll
  for (int j = 0; j < 4; ++j) v[j] = r[lane + j * 32];
  float mx = wred_max(fmaxf(fmaxf(v[0], v[1]), fmaxf(v[2], v[3])));
  float s = 0.f;
#pragma unroll
  for (int j = 0; j < 4; ++j) { v[j] = __expf(v[j] - mx); s += v[j]; }
  float inv = 1.0f / wred_sum(s);
#pragma unroll
  for (int j = 0; j < 4; ++j) AQ[(size_t)row * Sq + lane + j * 32] = f2bf(v[j] * inv);
}

// ---------------- AC[s][t] = softmax_t( Hm[t][s] ) (column softmax) ----------------
__global__ void __launch_bounds__(128) softmax_col(
    const float* __restrict__ Hm, unsigned short* __restrict__ AC) {
  int row = blockIdx.x * 4 + (threadIdx.x >> 5);  // b*128 + s
  int lane = threadIdx.x & 31;
  int bz = row >> 7, s_ = row & (Sq - 1);
  const float* base = Hm + (size_t)bz * (Sq * Sq) + s_;
  float v[4];
#pragma unroll
  for (int j = 0; j < 4; ++j) v[j] = base[(size_t)(lane + j * 32) * Sq];
  float mx = wred_max(fmaxf(fmaxf(v[0], v[1]), fmaxf(v[2], v[3])));
  float s = 0.f;
#pragma unroll
  for (int j = 0; j < 4; ++j) { v[j] = __expf(v[j] - mx); s += v[j]; }
  float inv = 1.0f / wred_sum(s);
#pragma unroll
  for (int j = 0; j < 4; ++j) AC[(size_t)row * Sq + lane + j * 32] = f2bf(v[j] * inv);
}

// ---------------- Out[z] = A[z] @ Bkn[z], per batch: M=128 N=512 K=128 ----------------
__global__ void __launch_bounds__(256) av_gemm(
    const unsigned short* __restrict__ Amat, const unsigned short* __restrict__ Bkn,
    float* __restrict__ OutF, unsigned short* __restrict__ OutB) {
  int z = blockIdx.z;
  int lane = threadIdx.x & 31, wid = threadIdx.x >> 5;
  int half = lane >> 4, l15 = lane & 15;
  int wm = wid >> 1, wn = wid & 1;
  int mBase = wm * 32;
  int nBase = blockIdx.y * 64 + wn * 32;
  const unsigned short* Ab = Amat + (size_t)z * (Sq * Sq);
  const unsigned short* Bz = Bkn + (size_t)z * Sq * H2;
  v8f acc[2][2];
#pragma unroll
  for (int i = 0; i < 2; ++i)
#pragma unroll
    for (int j = 0; j < 2; ++j) {
      v8f z8 = {0.f, 0.f, 0.f, 0.f, 0.f, 0.f, 0.f, 0.f};
      acc[i][j] = z8;
    }
  const unsigned short* a0r = Ab + (size_t)(mBase + l15) * Sq;
  const unsigned short* a1r = Ab + (size_t)(mBase + 16 + l15) * Sq;
#pragma unroll
  for (int kk = 0; kk < 4; ++kk) {
    int k0 = kk * 32;
    v16bf a0 = ld_frag_row(a0r, k0, half);
    v16bf a1 = ld_frag_row(a1r, k0, half);
    v16bf b0 = ld_frag_kn(Bz, H2, k0, nBase + l15, half);
    v16bf b1 = ld_frag_kn(Bz, H2, k0, nBase + 16 + l15, half);
    acc[0][0] = WMMA_BF16(a0, b0, acc[0][0]);
    acc[0][1] = WMMA_BF16(a0, b1, acc[0][1]);
    acc[1][0] = WMMA_BF16(a1, b0, acc[1][0]);
    acc[1][1] = WMMA_BF16(a1, b1, acc[1][1]);
  }
#pragma unroll
  for (int i = 0; i < 2; ++i)
#pragma unroll
    for (int j = 0; j < 2; ++j)
#pragma unroll
      for (int r = 0; r < 8; ++r) {
        int m = mBase + i * 16 + r + half * 8;
        int n = nBase + j * 16 + l15;
        float v = acc[i][j][r];
        OutF[(size_t)z * (Sq * H2) + (size_t)m * H2 + n] = v;
        if (OutB) OutB[((size_t)z * Sq + m) * H2 + n] = f2bf(v);
      }
}

// ---------------- host launch ----------------
extern "C" void kernel_launch(void* const* d_in, const int* in_sizes, int n_in,
                              void* d_out, int out_size, void* d_ws, size_t ws_size,
                              hipStream_t stream) {
  (void)in_sizes; (void)n_in; (void)out_size; (void)ws_size;
  const int*   inQ   = (const int*)d_in[1];
  const int*   inC   = (const int*)d_in[2];
  const int*   inC2  = (const int*)d_in[3];
  const float* hidQ  = (const float*)d_in[4];
  const float* celQ  = (const float*)d_in[5];
  const float* hidC  = (const float*)d_in[6];
  const float* celC  = (const float*)d_in[7];
  const float* hidC2 = (const float*)d_in[8];
  const float* celC2 = (const float*)d_in[9];
  const float* emb   = (const float*)d_in[10];
  const float* Wihf  = (const float*)d_in[11];
  const float* Whhf  = (const float*)d_in[12];
  const float* bihf  = (const float*)d_in[13];
  const float* bhhf  = (const float*)d_in[14];
  const float* Wihb  = (const float*)d_in[15];
  const float* Whhb  = (const float*)d_in[16];
  const float* bihb  = (const float*)d_in[17];
  const float* bhhb  = (const float*)d_in[18];
  const float* S1W   = (const float*)d_in[19];
  const float* S1b   = (const float*)d_in[20];
  float* out = (float*)d_out;

  char* ws = (char*)d_ws;
  size_t off = 0;
  auto take = [&](size_t bytes) -> char* {
    char* p = ws + off;
    off = (off + bytes + 255) & ~(size_t)255;
    return p;
  };
  unsigned short* Xall  = (unsigned short*)take((size_t)3 * MROWS * E * 2);
  unsigned short* Wcomb = (unsigned short*)take((size_t)2 * G4 * H2 * 2);
  float*          biasF = (float*)take((size_t)2 * G4 * 4);
  unsigned short* S1Wb  = (unsigned short*)take((size_t)H2 * H2 * 2);
  unsigned short* Yall  = (unsigned short*)take((size_t)3 * MROWS * H2 * 2);
  unsigned short* Ct    = (unsigned short*)take((size_t)MROWS * H2 * 2);
  float*          Hm    = (float*)take((size_t)Bb * Sq * Sq * 4);
  unsigned short* AQ    = (unsigned short*)take((size_t)Bb * Sq * Sq * 2);
  unsigned short* AC    = (unsigned short*)take((size_t)Bb * Sq * Sq * 2);
  unsigned short* MQb   = (unsigned short*)take((size_t)MROWS * H2 * 2);
  unsigned short* Hg    = (unsigned short*)take((size_t)6 * 2 * Bb * H * 2);
  unsigned int*   Bar   = (unsigned int*)take((size_t)6 * 16 * 4);

  hipMemsetAsync(Bar, 0, (size_t)6 * 16 * 4, stream);

  prep_kernel<<<512, 256, 0, stream>>>(Wihf, Whhf, Wihb, Whhb,
                                       bihf, bhhf, bihb, bhhb,
                                       S1W, Wcomb, biasF, S1Wb);
  gather_kernel<<<4096, 256, 0, stream>>>(inQ, inC, inC2, emb, Xall);

  size_t lds_bytes = (size_t)(128 * WSTR + 2 * 128 * XSTR) * sizeof(unsigned short);
  lstm_chain_kernel<<<6 * WPC, 256, lds_bytes, stream>>>(
      Xall, Wcomb, biasF, hidQ, celQ, hidC, celC, hidC2, celC2, Hg, Bar, Yall);

  const unsigned short* Yq = Yall;
  for (int a = 0; a < 2; ++a) {
    const unsigned short* Yc = Yall + (size_t)(1 + a) * MROWS * H2;
    float* outQ = out + (size_t)(2 * a) * MROWS * H2;      // MQ  -> attn_Q{1,2}
    float* outC = out + (size_t)(2 * a + 1) * MROWS * H2;  // MC  -> attn_C{,2}
    s1_gemm<<<dim3(128, 8), 256, 0, stream>>>(Yc, S1Wb, S1b, Ct);
    hm_gemm<<<dim3(1, 2, Bb), 256, 0, stream>>>(Yq, Ct, Hm);
    softmax_row<<<MROWS / 4, 128, 0, stream>>>(Hm, AQ);
    softmax_col<<<MROWS / 4, 128, 0, stream>>>(Hm, AC);
    av_gemm<<<dim3(1, 8, Bb), 256, 0, stream>>>(AQ, Yq, outQ, MQb);
    av_gemm<<<dim3(1, 8, Bb), 256, 0, stream>>>(AC, MQb, outC, nullptr);
  }
}
